// SCAGRU_22136261443923
// MI455X (gfx1250) — matrix operand order repaired and checked by reading, not compiled, not run on hardware
//
#include <hip/hip_runtime.h>

#define T_STEPS 12
#define BATCH   128
#define NDIM    2048
#define HID     64
#define ROWS    (BATCH * NDIM)   // 262144 independent GRU rows
#define WT_STRIDE 72             // padded LDS row stride (halfs)

typedef __attribute__((ext_vector_type(16))) _Float16 v16h;
typedef __attribute__((ext_vector_type(8)))  _Float16 v8h;
typedef __attribute__((ext_vector_type(8)))  float    v8f;

union V16U { v16h v; v8h h[2]; };

// v16h fragment = two 16B LDS reads: halfs [0..7] at p, [8..15] at p+off1.
__device__ __forceinline__ v16h ld_frag(const _Float16* p, int off1) {
  V16U u;
  u.h[0] = *(const v8h*)(p);
  u.h[1] = *(const v8h*)(p + off1);
  return u.v;
}

// 16x64 @ 64x16 via two chained v_wmma_f32_16x16x32_f16, C preloaded.
// B layout (16-bit 32x16): lanes0-15 K=0..7,16..23, lanes16-31 K=8..15,24..31 (matches A).
__device__ __forceinline__ v8f wmma_k64(v16h a0, v16h a1, const _Float16* wrow,
                                        int lhi, v8f c) {
  v16h b0 = ld_frag(wrow + lhi * 8, 16);
  c = __builtin_amdgcn_wmma_f32_16x16x32_f16(false, a0, false, b0, (short)0, c, false, false);
  v16h b1 = ld_frag(wrow + 32 + lhi * 8, 16);
  c = __builtin_amdgcn_wmma_f32_16x16x32_f16(false, a1, false, b1, (short)0, c, false, false);
  return c;
}

__global__ __launch_bounds__(256) __attribute__((amdgpu_waves_per_eu(4)))
void scagru_kernel(
    const float* __restrict__ input, const float* __restrict__ state,
    const float* __restrict__ W_zh, const float* __restrict__ W_zx, const float* __restrict__ b_z,
    const float* __restrict__ W_rh, const float* __restrict__ W_rx, const float* __restrict__ b_r,
    const float* __restrict__ W_hh, const float* __restrict__ W_hx, const float* __restrict__ b_h,
    const float* __restrict__ W_ch, const float* __restrict__ b_c,
    float* __restrict__ out)
{
  __shared__ __align__(16) _Float16 sW[4][64 * WT_STRIDE];   // 36 KB: W^T f16
  __shared__ __align__(16) _Float16 sH[8][16 * WT_STRIDE];   // 18 KB: per-wave transpose scratch
  __shared__ float sX[8][16];                                // x broadcast

  const int tid  = threadIdx.x;
  const int lane = tid & 31;
  const int wave = tid >> 5;
  const int llo  = lane & 15;
  const int lhi  = lane >> 4;
  const int rowbase = (blockIdx.x * 8 + wave) * 16;

  // ---- stage weights (W^T[n][k], f16) ----
  const float* wsrc[4] = { W_zh, W_rh, W_hh, W_ch };
#pragma unroll
  for (int m = 0; m < 4; ++m)
    for (int i = tid; i < 64 * 64; i += 256) {
      int k = i >> 6, n = i & 63;
      sW[m][n * WT_STRIDE + k] = (_Float16)wsrc[m][i];
    }
  __syncthreads();

  // per-lane per-tile scalars
  float wzx[4], wrx[4], whx[4], bz[4], br[4], bh[4];
#pragma unroll
  for (int t = 0; t < 4; ++t) {
    const int f = t * 16 + llo;
    wzx[t] = W_zx[f]; wrx[t] = W_rx[f]; whx[t] = W_hx[f];
    bz[t]  = b_z[f];  br[t]  = b_r[f];  bh[t]  = b_h[f];
  }

  // ---- h = state in C layout: vgpr r, tile t -> row r+lhi*8, col t*16+llo ----
  v8f hC[4];
#pragma unroll
  for (int t = 0; t < 4; ++t)
#pragma unroll
    for (int r = 0; r < 8; ++r)
      hC[t][r] = state[(size_t)(rowbase + r + lhi * 8) * HID + t * 16 + llo];

  _Float16* myH = &sH[wave][0];

  auto stage = [&](const v8f* src) {        // C-layout regs -> row-major f16 LDS tile
#pragma unroll
    for (int t = 0; t < 4; ++t)
#pragma unroll
      for (int r = 0; r < 8; ++r)
        myH[(r + lhi * 8) * WT_STRIDE + t * 16 + llo] = (_Float16)src[t][r];
  };

  // A layout (16-bit 16x32): lane row = llo; two 16B chunks 32B apart per K-tile.
  auto loadA0 = [&]() { return ld_frag(myH + llo * WT_STRIDE + lhi * 8, 16); };
  auto loadA1 = [&]() { return ld_frag(myH + llo * WT_STRIDE + lhi * 8 + 32, 16); };

  // ---- c = relu(state @ W_ch + b_c), kept as single C-layout array ----
  stage(hC);
  asm volatile("s_wait_dscnt 0x0" ::: "memory");
  v16h a0 = loadA0(), a1 = loadA1();
  v8f cC[4];
#pragma unroll
  for (int t = 0; t < 4; ++t) {
    const float bcv = b_c[t * 16 + llo];
    v8f pre;
#pragma unroll
    for (int i = 0; i < 8; ++i) pre[i] = bcv;
    v8f acc = wmma_k64(a0, a1, &sW[3][(t * 16 + llo) * WT_STRIDE], lhi, pre);
#pragma unroll
    for (int i = 0; i < 8; ++i) cC[t][i] = acc[i] > 0.f ? acc[i] : 0.f;
  }

  // ---- recurrence ----
  for (int step = 0; step < T_STEPS; ++step) {
    const float* xt = input + (size_t)step * ROWS + rowbase;
    if (lane < 16) sX[wave][lane] = xt[lane];
    stage(hC);
    asm volatile("s_wait_dscnt 0x0" ::: "memory");
    a0 = loadA0(); a1 = loadA1();                 // h fragments (A operand)
    float xr[8];
#pragma unroll
    for (int r = 0; r < 8; ++r) xr[r] = sX[wave][r + lhi * 8];

    // r gate, consumed immediately into h*r (no persistent r array)
    v8f hr[4];
#pragma unroll
    for (int t = 0; t < 4; ++t) {
      v8f pre;
#pragma unroll
      for (int i = 0; i < 8; ++i) pre[i] = (cC[t][i] + br[t]) + xr[i] * wrx[t];
      v8f ar = wmma_k64(a0, a1, &sW[1][(t * 16 + llo) * WT_STRIDE], lhi, pre);
#pragma unroll
      for (int i = 0; i < 8; ++i)
        hr[t][i] = hC[t][i] * (1.f / (1.f + __expf(-ar[i])));
    }

    stage(hr);                                    // h frags already live in a0/a1
    asm volatile("s_wait_dscnt 0x0" ::: "memory");
    const v16h g0 = loadA0(), g1 = loadA1();      // (h*r) fragments

    // z + candidate fused per tile: no persistent z array
#pragma unroll
    for (int t = 0; t < 4; ++t) {
      v8f preZ, preH;
#pragma unroll
      for (int i = 0; i < 8; ++i) {
        preZ[i] = (cC[t][i] + bz[t]) + xr[i] * wzx[t];
        preH[i] = (cC[t][i] + bh[t]) + xr[i] * whx[t];
      }
      v8f az = wmma_k64(a0, a1, &sW[0][(t * 16 + llo) * WT_STRIDE], lhi, preZ);
      v8f ah = wmma_k64(g0, g1, &sW[2][(t * 16 + llo) * WT_STRIDE], lhi, preH);
#pragma unroll
      for (int i = 0; i < 8; ++i) {
        const float z  = 1.f / (1.f + __expf(-az[i]));
        const float ht = 2.f / (1.f + __expf(-2.f * ah[i])) - 1.f;  // tanh
        hC[t][i] = hC[t][i] + z * (ht - hC[t][i]);
      }
    }
  }

  // ---- write h (coalesced C-layout pattern) ----
#pragma unroll
  for (int t = 0; t < 4; ++t)
#pragma unroll
    for (int r = 0; r < 8; ++r)
      out[(size_t)(rowbase + r + lhi * 8) * HID + t * 16 + llo] = hC[t][r];
}

extern "C" void kernel_launch(void* const* d_in, const int* in_sizes, int n_in,
                              void* d_out, int out_size, void* d_ws, size_t ws_size,
                              hipStream_t stream) {
  (void)in_sizes; (void)n_in; (void)out_size; (void)d_ws; (void)ws_size;
  const float* input = (const float*)d_in[0];
  const float* state = (const float*)d_in[1];
  const float* W_zh  = (const float*)d_in[2];
  const float* W_zx  = (const float*)d_in[3];
  const float* b_z   = (const float*)d_in[4];
  const float* W_rh  = (const float*)d_in[5];
  const float* W_rx  = (const float*)d_in[6];
  const float* b_r   = (const float*)d_in[7];
  const float* W_hh  = (const float*)d_in[8];
  const float* W_hx  = (const float*)d_in[9];
  const float* b_h   = (const float*)d_in[10];
  const float* W_ch  = (const float*)d_in[11];
  const float* b_c   = (const float*)d_in[12];
  float* out = (float*)d_out;

  dim3 grid(ROWS / 128), block(256);   // 2048 blocks x 8 wave32
  scagru_kernel<<<grid, block, 0, stream>>>(input, state,
      W_zh, W_zx, b_z, W_rh, W_rx, b_r, W_hh, W_hx, b_h, W_ch, b_c, out);
}
